// AEFIT_43550968381956
// MI455X (gfx1250) — matrix operand order repaired and checked by reading, not compiled, not run on hardware
//
#include <hip/hip_runtime.h>

typedef unsigned short u16;
typedef unsigned int   u32;

typedef __attribute__((ext_vector_type(16))) __bf16 v16bf;
typedef __attribute__((ext_vector_type(8)))  float  v8f;

union FragBF { v16bf v; uint4 q[2]; };

constexpr int Bn = 65536;
constexpr int Ln = 512;
constexpr int Vn = 128;
constexpr int Dn = 1024;
constexpr float LOG2PI_F = 1.8378770664093453f;

// ---- workspace layout (bytes) ----
constexpr size_t ACC_OFF = 0;                                   // 4 floats
constexpr size_t W2P_OFF = 256;                                 // 1024*1024 bf16 = 2 MB
constexpr size_t W3P_OFF = W2P_OFF + (size_t)Dn * Dn * 2;       // 1024*256 bf16 = 512 KB
constexpr size_t Y_OFF   = W3P_OFF + (size_t)Dn * 2 * Vn * 2;   // B*D bf16 = 128 MB
constexpr size_t H_OFF   = Y_OFF + (size_t)Bn * Dn * 2;         // B*D bf16 = 128 MB
constexpr size_t ENC_OFF = H_OFF + (size_t)Bn * Dn * 2;         // B*256 f32 = 64 MB

__device__ __forceinline__ u16 f2bf(float f) {
  u32 x = __float_as_uint(f);
  u32 r = x + 0x7fffu + ((x >> 16) & 1u);   // round-to-nearest-even
  return (u16)(r >> 16);
}

__device__ __forceinline__ float wred(float v) {
#pragma unroll
  for (int o = 16; o > 0; o >>= 1) v += __shfl_down(v, o, 32);
  return v;
}

// ---------------- weight packing into WMMA B-fragment order ----------------
// B-matrix (32x16, KxN) bf16 fragment order used here (contiguous k, same as A): lane = khalf*16 + n%16,
// lanes 0-15 hold K=0..15 in dwords 0..7 (pairs), lanes 16-31 hold K=16..31.
__global__ void pack_w2_kernel(const float* __restrict__ W2, u16* __restrict__ W2p) {
  int t = blockIdx.x * 256 + threadIdx.x;   // < 32*64*32*8 = 524288 dwords
  int d    = t & 7;
  int lane = (t >> 3) & 31;
  int nt   = (t >> 8) & 63;
  int kt   = t >> 14;
  int n = nt * 16 + (lane & 15);
  int k = kt * 32 + (lane >> 4) * 16 + 2 * d;
  u32 lo = f2bf(W2[(size_t)k * Dn + n]);
  u32 hi = f2bf(W2[(size_t)(k + 1) * Dn + n]);
  ((u32*)W2p)[t] = lo | (hi << 16);
}

__global__ void pack_w3_kernel(const float* __restrict__ W3, u16* __restrict__ W3p) {
  int t = blockIdx.x * 256 + threadIdx.x;   // < 32*16*32*8 = 131072 dwords
  int d    = t & 7;
  int lane = (t >> 3) & 31;
  int nt   = (t >> 8) & 15;
  int kt   = t >> 12;
  int n = nt * 16 + (lane & 15);            // 0..255
  int k = kt * 32 + (lane >> 4) * 16 + 2 * d;
  u32 lo = f2bf(W3[(size_t)k * 256 + n]);
  u32 hi = f2bf(W3[(size_t)(k + 1) * 256 + n]);
  ((u32*)W3p)[t] = lo | (hi << 16);
}

// ---------------- stable compaction of finite values + scale by w1 --------
__global__ void __launch_bounds__(256) compact_kernel(
    const float* __restrict__ xy, const float* __restrict__ w1,
    const float* __restrict__ b1, u16* __restrict__ Y) {
  int lane = threadIdx.x & 31;
  int w    = threadIdx.x >> 5;
  int row  = blockIdx.x * 8 + w;
  const float* xrow = xy + (size_t)row * Dn;
  u16* yrow = Y + (size_t)row * Dn;
  int base = 0;
  for (int ch = 0; ch < Dn / 32; ++ch) {
    int c = ch * 32 + lane;
    float v = xrow[c];
    bool fin = ((__float_as_uint(v) & 0x7f800000u) != 0x7f800000u);
    u32 mask = __builtin_amdgcn_ballot_w32(fin);
    int rank = __popc(mask & ((1u << lane) - 1u));
    if (fin) {
      int pos = base + rank;
      yrow[pos] = f2bf(v * w1[pos] + b1[pos]);
    }
    base += __popc(mask);
  }
  for (int p = base + lane; p < Dn; p += 32) yrow[p] = 0;
}

// ---------------- GEMM1: H = relu(Y @ W2 + b2), bf16 WMMA -----------------
// block = 128 rows; full Y slab (128x1024 bf16, padded) is DMA'd into LDS
// with GLOBAL_LOAD_ASYNC_TO_LDS_B128 (ASYNCcnt path), then 8 N-blocks of 128
// are computed so Y is read from HBM exactly once.
__global__ void __launch_bounds__(256) gemm1_kernel(
    const u16* __restrict__ Y, const u16* __restrict__ W2p,
    const float* __restrict__ b2, u16* __restrict__ H) {
  extern __shared__ u16 slab[];             // 128 x 1032 (pad 8 bf16 / row)
  const int LSTR = 1032;
  int tid = threadIdx.x;
  int mb = blockIdx.x;
  const uint4* ysrc = (const uint4*)(Y + (size_t)mb * 128 * Dn);
  for (int i = tid; i < 128 * 128; i += 256) {   // 128 uint4 per row
    int row = i >> 7;
    int c16 = i & 127;
    unsigned long long gaddr =
        (unsigned long long)(uintptr_t)(const void*)(ysrc + row * 128 + c16);
    // low 32 bits of a flat LDS pointer == LDS byte offset (VDST operand)
    unsigned ldsoff = (unsigned)(uintptr_t)(const void*)(slab + row * LSTR + c16 * 8);
    asm volatile("global_load_async_to_lds_b128 %0, %1, off"
                 :: "v"(ldsoff), "v"(gaddr) : "memory");
  }
  asm volatile("s_wait_asynccnt 0" ::: "memory");
  __syncthreads();
  int lane = tid & 31;
  int wv = tid >> 5;
  int mwave = wv & 3, nwave = wv >> 2;      // 4 waves over M, 2 over N
  int lrow = lane & 15, khalf = lane >> 4;
  int m0 = mwave * 32;
  for (int nb = 0; nb < 8; ++nb) {
    v8f acc[2][4];
#pragma unroll
    for (int i = 0; i < 2; ++i)
#pragma unroll
      for (int j = 0; j < 4; ++j)
#pragma unroll
        for (int e = 0; e < 8; ++e) acc[i][j][e] = 0.0f;
    int ntbase = nb * 8 + nwave * 4;
    for (int kt = 0; kt < 32; ++kt) {
      FragBF a[2], b[4];
#pragma unroll
      for (int i = 0; i < 2; ++i) {
        const u16* p = slab + (m0 + i * 16 + lrow) * LSTR + kt * 32 + khalf * 16;
        a[i].q[0] = *(const uint4*)p;
        a[i].q[1] = *(const uint4*)(p + 8);
      }
#pragma unroll
      for (int j = 0; j < 4; ++j) {
        const uint4* p =
            (const uint4*)(W2p + ((size_t)((kt * 64 + ntbase + j) * 32 + lane)) * 16);
        b[j].q[0] = p[0];
        b[j].q[1] = p[1];
      }
#pragma unroll
      for (int i = 0; i < 2; ++i)
#pragma unroll
        for (int j = 0; j < 4; ++j)
          acc[i][j] = __builtin_amdgcn_wmma_f32_16x16x32_bf16(
              false, a[i].v, false, b[j].v, (short)0, acc[i][j], false, false);
    }
    int rowg0 = mb * 128 + m0;
#pragma unroll
    for (int i = 0; i < 2; ++i) {
#pragma unroll
      for (int j = 0; j < 4; ++j) {
        int ncol = nb * 128 + nwave * 64 + j * 16 + lrow;
        float bias = b2[ncol];
#pragma unroll
        for (int e = 0; e < 8; ++e) {
          int mr = rowg0 + i * 16 + e + 8 * khalf;
          float val = fmaxf(acc[i][j][e] + bias, 0.0f);
          H[(size_t)mr * Dn + ncol] = f2bf(val);
        }
      }
    }
  }
}

// ---------------- GEMM2: ENC = H @ W3 + b3 (f32 out), bf16 WMMA -----------
__global__ void __launch_bounds__(256) gemm2_kernel(
    const u16* __restrict__ H, const u16* __restrict__ W3p,
    const float* __restrict__ b3, float* __restrict__ enc) {
  int tid = threadIdx.x;
  int mb   = blockIdx.x >> 1;
  int nblk = blockIdx.x & 1;
  int lane = tid & 31;
  int wv = tid >> 5;
  int mwave = wv & 3, nwave = wv >> 2;
  int lrow = lane & 15, khalf = lane >> 4;
  int m0 = mwave * 32;
  v8f acc[2][4];
#pragma unroll
  for (int i = 0; i < 2; ++i)
#pragma unroll
    for (int j = 0; j < 4; ++j)
#pragma unroll
      for (int e = 0; e < 8; ++e) acc[i][j][e] = 0.0f;
  int ntbase = nblk * 8 + nwave * 4;
  for (int kt = 0; kt < 32; ++kt) {
    FragBF a[2], b[4];
#pragma unroll
    for (int i = 0; i < 2; ++i) {
      const u16* p = H + (size_t)(mb * 128 + m0 + i * 16 + lrow) * Dn + kt * 32 + khalf * 16;
      a[i].q[0] = *(const uint4*)p;
      a[i].q[1] = *(const uint4*)(p + 8);
    }
#pragma unroll
    for (int j = 0; j < 4; ++j) {
      const uint4* p =
          (const uint4*)(W3p + ((size_t)((kt * 16 + ntbase + j) * 32 + lane)) * 16);
      b[j].q[0] = p[0];
      b[j].q[1] = p[1];
    }
#pragma unroll
    for (int i = 0; i < 2; ++i)
#pragma unroll
      for (int j = 0; j < 4; ++j)
        acc[i][j] = __builtin_amdgcn_wmma_f32_16x16x32_bf16(
            false, a[i].v, false, b[j].v, (short)0, acc[i][j], false, false);
  }
  int rowg0 = mb * 128 + m0;
#pragma unroll
  for (int i = 0; i < 2; ++i) {
#pragma unroll
    for (int j = 0; j < 4; ++j) {
      int ncol = nblk * 128 + nwave * 64 + j * 16 + lrow;
      float bias = b3[ncol];
#pragma unroll
      for (int e = 0; e < 8; ++e) {
        int mr = rowg0 + i * 16 + e + 8 * khalf;
        enc[(size_t)mr * 256 + ncol] = acc[i][j][e] + bias;
      }
    }
  }
}

// ---------------- tail: sample, decoder, losses (wave per row) ------------
__global__ void __launch_bounds__(256) tail_kernel(
    const float* __restrict__ enc, const float* __restrict__ eps,
    const float* __restrict__ xy, const int* __restrict__ att,
    const float* __restrict__ Wg1, const float* __restrict__ bg1,
    const float* __restrict__ Wg2, const float* __restrict__ bg2,
    float* __restrict__ accs) {
  __shared__ float s_s[8][Vn];
  __shared__ float s_d1[8][16];
  int lane = threadIdx.x & 31;
  int w    = threadIdx.x >> 5;
  int row  = blockIdx.x * 8 + w;
  const float* erow = enc + (size_t)row * 256;
  float logpz = 0.0f, logqz = 0.0f;
  for (int v = lane; v < Vn; v += 32) {
    float mean = erow[v];
    float lg   = erow[Vn + v];
    float e    = eps[(size_t)row * Vn + v];
    float sv   = e * __expf(0.5f * lg) + mean;
    s_s[w][v] = sv;
    logpz += -0.5f * (sv * sv + LOG2PI_F);
    // (s-mean)^2 * exp(-logv) == eps^2
    logqz += -0.5f * (e * e + lg + LOG2PI_F);
  }
  __syncthreads();
  if (lane < 10) {
    float a = bg1[lane];
    for (int v = 0; v < Vn; ++v) a += s_s[w][v] * Wg1[v * 10 + lane];
    s_d1[w][lane] = fmaxf(a, 0.0f);
  }
  __syncthreads();
  float l_ce = 0.0f, l_sq = 0.0f, l_den = 0.0f;
  const float* xrow = xy + (size_t)row * Dn;
  const int*   arow = att + (size_t)row * Ln;
  for (int c = lane; c < Dn; c += 32) {
    float xv = xrow[c];
    bool fin = ((__float_as_uint(xv) & 0x7f800000u) != 0x7f800000u);
    float xc = fin ? xv : 0.0f;
    float af = (float)arow[c & (Ln - 1)];
    float XY = bg2[c];
#pragma unroll
    for (int u = 0; u < 10; ++u) XY += s_d1[w][u] * Wg2[u * Dn + c];
    float ce = fmaxf(XY, 0.0f) - XY * xc + log1pf(__expf(-fabsf(XY)));
    l_ce += ce * af;
    float d = xc - XY;
    l_sq += 0.5f * d * d * af;
    if (c < Ln) l_den += af;
  }
  l_ce  = wred(l_ce);
  l_sq  = wred(l_sq);
  l_den = wred(l_den);
  logpz = wred(logpz);
  logqz = wred(logqz);
  if (lane == 0) {
    // l_vae contribution: mean over rows of (sum_ce - logpz + logqz)
    atomicAdd(&accs[0], (l_ce - logpz + logqz) * (1.0f / Bn));
    atomicAdd(&accs[1], l_sq);
    atomicAdd(&accs[2], l_den);
  }
}

__global__ void zero_acc_kernel(float* accs) {
  if (threadIdx.x < 4) accs[threadIdx.x] = 0.0f;
}

__global__ void finalize_kernel(const float* __restrict__ accs, float* __restrict__ out) {
  out[0] = accs[0] + __expf(accs[1] / accs[2]);
}

extern "C" void kernel_launch(void* const* d_in, const int* in_sizes, int n_in,
                              void* d_out, int out_size, void* d_ws, size_t ws_size,
                              hipStream_t stream) {
  const float* xy  = (const float*)d_in[0];
  const int*   att = (const int*)d_in[1];
  const float* eps = (const float*)d_in[2];
  const float* w1  = (const float*)d_in[3];
  const float* b1  = (const float*)d_in[4];
  const float* W2  = (const float*)d_in[5];
  const float* b2  = (const float*)d_in[6];
  const float* W3  = (const float*)d_in[7];
  const float* b3  = (const float*)d_in[8];
  const float* Wg1 = (const float*)d_in[9];
  const float* bg1 = (const float*)d_in[10];
  const float* Wg2 = (const float*)d_in[11];
  const float* bg2 = (const float*)d_in[12];

  char* ws = (char*)d_ws;
  float* accs = (float*)(ws + ACC_OFF);
  u16* w2p   = (u16*)(ws + W2P_OFF);
  u16* w3p   = (u16*)(ws + W3P_OFF);
  u16* Yb    = (u16*)(ws + Y_OFF);
  u16* Hb    = (u16*)(ws + H_OFF);
  float* encb = (float*)(ws + ENC_OFF);

  zero_acc_kernel<<<1, 32, 0, stream>>>(accs);
  pack_w2_kernel<<<524288 / 256, 256, 0, stream>>>(W2, w2p);
  pack_w3_kernel<<<131072 / 256, 256, 0, stream>>>(W3, w3p);
  compact_kernel<<<Bn / 8, 256, 0, stream>>>(xy, w1, b1, Yb);
  gemm1_kernel<<<Bn / 128, 256, 128 * 1032 * 2, stream>>>(Yb, w2p, b2, Hb);
  gemm2_kernel<<<(Bn / 128) * 2, 256, 0, stream>>>(Hb, w3p, b3, encb);
  tail_kernel<<<Bn / 8, 256, 0, stream>>>(encb, eps, xy, att, Wg1, bg1, Wg2, bg2, accs);
  finalize_kernel<<<1, 1, 0, stream>>>(accs, (float*)d_out);
}
